// GVPAttentionShortBranch_19567871001000
// MI455X (gfx1250) — matrix operand order repaired and checked by reading, not compiled, not run on hardware
//
#include <hip/hip_runtime.h>
#include <hip/hip_bf16.h>
#include <math.h>

// ---------------- problem constants ----------------
#define NN 5000           // nodes
#define NE 80000          // edges
#define NCONF 2
#define NC (NN * NCONF)   // node-confs = 10000
#define EC (NE * NCONF)   // edge-confs = 160000

typedef __attribute__((ext_vector_type(8)))  float  v8f;
typedef __attribute__((ext_vector_type(16))) __bf16 v16bf;

struct alignas(16) U4 { unsigned x, y, z, w; };
union Frag { v16bf v; U4 q[2]; };

__device__ __forceinline__ unsigned short f2bf(float f) {
  unsigned u = __float_as_uint(f);
  u += 0x7FFFu + ((u >> 16) & 1u);      // round-to-nearest-even
  return (unsigned short)(u >> 16);
}
__device__ __forceinline__ float siluf(float x) { return x / (1.f + __expf(-x)); }

__device__ float atomicMaxF(float* addr, float val) {
  int* ia = (int*)addr;
  int old = __float_as_int(*addr);
  while (__int_as_float(old) < val) {
    int assumed = old;
    old = atomicCAS(ia, assumed, __float_as_int(val));
    if (old == assumed) break;
  }
  return __int_as_float(old);
}

// ---------------- WMMA GEMM: C[MxN] = bf16(A[MxKp]) @ Wt[NxKp]^T + bias ----------------
// One wave computes a 16 x (NT*16) tile. Requires gridDim.y * NT * 16 == N, so the
// K-loop is branch-free (no accumulator spills). Rows >= M are computed on a clamped
// A row and simply never stored (A row m only contributes to D row m).
template <int NT>
__global__ void k_gemm_bf16(const unsigned short* __restrict__ A,
                            const unsigned short* __restrict__ Wt,
                            const float* __restrict__ bias,
                            float* __restrict__ C,
                            int M, int N, int Kp) {
  const int lane = threadIdx.x & 31;
  const int half = lane >> 4;
  const int l16  = lane & 15;
  const int mt   = blockIdx.x;
  const int nt0  = blockIdx.y * NT;
  const int m    = mt * 16 + l16;
  const int mc   = (m < M) ? m : 0;                 // clamp; guarded at store
  const unsigned short* ap  = A  + (size_t)mc * Kp + 8 * half;
  const unsigned short* bp0 = Wt + (size_t)(nt0 * 16 + l16) * Kp + 8 * half;
  v8f acc[NT] = {};
  for (int kc = 0; kc < Kp; kc += 32) {
    Frag a;
    a.q[0] = *(const U4*)(ap + kc);        // K = kc + 8*half + [0..7]
    a.q[1] = *(const U4*)(ap + kc + 16);   // K = kc + 16 + 8*half + [0..7]
    __builtin_prefetch(ap + kc + 32, 0, 0);
#pragma unroll
    for (int t = 0; t < NT; ++t) {
      Frag b;
      const unsigned short* bp = bp0 + (size_t)t * 16 * Kp + kc;
      b.q[0] = *(const U4*)(bp);           // K = kc + 8*half + [0..7]
      b.q[1] = *(const U4*)(bp + 16);      // K = kc + 16 + 8*half + [0..7]
      acc[t] = __builtin_amdgcn_wmma_f32_16x16x32_bf16(
          false, a.v, false, b.v, (short)0, acc[t], false, false);
    }
  }
#pragma unroll
  for (int t = 0; t < NT; ++t) {
    int col = (nt0 + t) * 16 + l16;        // always < N by construction
    float bv = bias[col];
#pragma unroll
    for (int r = 0; r < 8; ++r) {
      int row = mt * 16 + r + 8 * half;    // C layout: M = r + 8*half
      if (row < M) C[(size_t)row * N + col] = acc[t][r] + bv;
    }
  }
}

// ---------------- weight convert: W[KxN] fp32 -> Wt[NxKp] bf16 (transposed, K-padded) ----------------
__global__ void k_w2bf(const float* __restrict__ W, unsigned short* __restrict__ Wt,
                       int K, int N, int Kp) {
  long idx = (long)blockIdx.x * blockDim.x + threadIdx.x;
  int k = (int)(idx % Kp); long n = idx / Kp;
  if (n >= N) return;
  float v = (k < K) ? W[(size_t)k * N + n] : 0.f;
  Wt[(size_t)n * Kp + k] = f2bf(v);
}

// ---------------- gvp_ln: scalar layernorm + vector norm (one wave per row) ----------------
__global__ void k_ln(const float* __restrict__ s, const float* __restrict__ v,
                     const float* __restrict__ g, const float* __restrict__ b,
                     float* __restrict__ ns, float* __restrict__ nv,
                     long rows, int S, int V) {
  long wid = ((long)blockIdx.x * blockDim.x + threadIdx.x) >> 5;
  int lane = threadIdx.x & 31;
  if (wid >= rows) return;
  const float* sr = s + (size_t)wid * S;
  float sum = 0.f, sq = 0.f;
  for (int i = lane; i < S; i += 32) { float x = sr[i]; sum += x; sq += x * x; }
  for (int o = 16; o > 0; o >>= 1) { sum += __shfl_xor(sum, o, 32); sq += __shfl_xor(sq, o, 32); }
  float mu = sum / S;
  float var = fmaxf(sq / S - mu * mu, 0.f);
  float inv = rsqrtf(var + 1e-5f);
  float* no = ns + (size_t)wid * S;
  for (int i = lane; i < S; i += 32) no[i] = (sr[i] - mu) * inv * g[i] + b[i];
  const float* vr = v + (size_t)wid * V * 3;
  float vs = 0.f;
  for (int i = lane; i < V * 3; i += 32) { float x = vr[i]; vs += x * x; }
  for (int o = 16; o > 0; o >>= 1) vs += __shfl_xor(vs, o, 32);
  float rv = 1.f / sqrtf(fmaxf(vs / V, 1e-8f));
  float* vo = nv + (size_t)wid * V * 3;
  for (int i = lane; i < V * 3; i += 32) vo[i] = vr[i] * rv;
}

// ---------------- generic A-tile builder: cols [0,S)=s, [S,Kp)=0 (norms overwritten later) ----
__global__ void k_build_A(const float* __restrict__ s, unsigned short* __restrict__ A,
                          int S, int Kp, long rows) {
  long idx = (long)blockIdx.x * blockDim.x + threadIdx.x;
  int c = (int)(idx % Kp); long r = idx / Kp;
  if (r >= rows) return;
  float v = (c < S) ? s[(size_t)r * S + c] : 0.f;
  A[(size_t)r * Kp + c] = f2bf(v);
}

// ---------------- vh = v @ wh ; write vh (fp32) and |vh| into A at Acol ----------------
__global__ void k_vh_norms(const float* __restrict__ v, const float* __restrict__ wh,
                           float* __restrict__ vh, unsigned short* __restrict__ A,
                           int Acol, int Kp, int VI, int H, long rows) {
  long idx = (long)blockIdx.x * blockDim.x + threadIdx.x;
  int h = (int)(idx % H); long r = idx / H;
  if (r >= rows) return;
  const float* vr = v + (size_t)r * VI * 3;
  float x = 0.f, y = 0.f, z = 0.f;
  for (int k = 0; k < VI; ++k) {
    float w = wh[(size_t)k * H + h];
    x = fmaf(vr[k * 3 + 0], w, x); y = fmaf(vr[k * 3 + 1], w, y); z = fmaf(vr[k * 3 + 2], w, z);
  }
  float* o = vh + ((size_t)r * H + h) * 3;
  o[0] = x; o[1] = y; o[2] = z;
  A[(size_t)r * Kp + Acol + h] = f2bf(sqrtf(fmaxf(x * x + y * y + z * z, 1e-8f)));
}

// ---------------- v_out = (vh @ wv) * sigmoid(s_pre @ wsv + bsv) ----------------
__global__ void k_gate(const float* __restrict__ spre, const float* __restrict__ vhb,
                       const float* __restrict__ wv, const float* __restrict__ wsv,
                       const float* __restrict__ bsv, float* __restrict__ vout,
                       int SO, int H, int VO, long rows) {
  long idx = (long)blockIdx.x * blockDim.x + threadIdx.x;
  int k = (int)(idx % VO); long r = idx / VO;
  if (r >= rows) return;
  const float* sr = spre + (size_t)r * SO;
  float d = bsv[k];
  for (int i = 0; i < SO; ++i) d = fmaf(sr[i], wsv[(size_t)i * VO + k], d);
  float g = 1.f / (1.f + __expf(-d));
  const float* vr = vhb + (size_t)r * H * 3;
  float x = 0.f, y = 0.f, z = 0.f;
  for (int h = 0; h < H; ++h) {
    float w = wv[(size_t)h * VO + k];
    x = fmaf(vr[h * 3 + 0], w, x); y = fmaf(vr[h * 3 + 1], w, y); z = fmaf(vr[h * 3 + 2], w, z);
  }
  float* o = vout + ((size_t)r * VO + k) * 3;
  o[0] = x * g; o[1] = y * g; o[2] = z * g;
}

__global__ void k_silu(float* __restrict__ p, long n) {
  long i = (long)blockIdx.x * blockDim.x + threadIdx.x;
  if (i < n) p[i] = siluf(p[i]);
}
__global__ void k_fill(float* __restrict__ p, float v, long n) {
  long i = (long)blockIdx.x * blockDim.x + threadIdx.x;
  if (i < n) p[i] = v;
}
__global__ void k_axpy(float* __restrict__ y, const float* __restrict__ x, long n) {
  long i = (long)blockIdx.x * blockDim.x + threadIdx.x;
  if (i < n) y[i] += x[i];
}

// ---------------- encoder message gather (scalar part): [ns[src] | es | ns[dst]] ----------------
__global__ void k_scat_enc(const int* __restrict__ EI, const float* __restrict__ NSb,
                           const float* __restrict__ ES, unsigned short* __restrict__ A) {
  long idx = (long)blockIdx.x * blockDim.x + threadIdx.x;
  int c = (int)(idx % 352); long ec = idx / 352;
  if (ec >= EC) return;
  long e = ec >> 1; int cf = (int)(ec & 1);
  float val = 0.f;
  if (c < 288) {
    int src = EI[e], dst = EI[NE + e];
    if (c < 128)      val = NSb[((size_t)src * 2 + cf) * 128 + c];
    else if (c < 160) val = ES[(size_t)ec * 32 + (c - 128)];
    else              val = NSb[((size_t)dst * 2 + cf) * 128 + (c - 160)];
  }
  A[(size_t)ec * 352 + c] = f2bf(val);
}

// encoder message: vh over concat v = [nv[src](16) | ev(1) | nv[dst](16)], wh 33x33, norms at col 288
__global__ void k_vh_msg_enc(const int* __restrict__ EI, const float* __restrict__ NVb,
                             const float* __restrict__ EV, const float* __restrict__ wh,
                             float* __restrict__ vh, unsigned short* __restrict__ A) {
  long idx = (long)blockIdx.x * blockDim.x + threadIdx.x;
  int h = (int)(idx % 33); long ec = idx / 33;
  if (ec >= EC) return;
  long e = ec >> 1; int cf = (int)(ec & 1);
  int src = EI[e], dst = EI[NE + e];
  float x = 0.f, y = 0.f, z = 0.f;
  for (int k = 0; k < 33; ++k) {
    const float* p;
    if (k < 16)       p = NVb + ((size_t)src * 2 + cf) * 48 + k * 3;
    else if (k == 16) p = EV + (size_t)ec * 3;
    else              p = NVb + ((size_t)dst * 2 + cf) * 48 + (k - 17) * 3;
    float w = wh[k * 33 + h];
    x = fmaf(p[0], w, x); y = fmaf(p[1], w, y); z = fmaf(p[2], w, z);
  }
  float* o = vh + ((size_t)ec * 33 + h) * 3;
  o[0] = x; o[1] = y; o[2] = z;
  A[(size_t)ec * 352 + 288 + h] = f2bf(sqrtf(fmaxf(x * x + y * y + z * z, 1e-8f)));
}

// ---------------- decoder message gathers (fwd = src<dst selects current vs encoder feats) -----
__global__ void k_scat_dec(const int* __restrict__ EI, const float* __restrict__ NSb,
                           const float* __restrict__ AES, const float* __restrict__ ESD,
                           unsigned short* __restrict__ A) {
  long idx = (long)blockIdx.x * blockDim.x + threadIdx.x;
  int c = (int)(idx % 352); long e = idx / 352;
  if (e >= NE) return;
  int src = EI[e], dst = EI[NE + e];
  const float* sb = (src < dst) ? NSb : AES;
  float val = 0.f;
  if (c < 128)      val = sb[(size_t)src * 128 + c];
  else if (c < 164) val = ESD[(size_t)e * 36 + (c - 128)];
  else if (c < 292) val = sb[(size_t)dst * 128 + (c - 164)];
  A[(size_t)e * 352 + c] = f2bf(val);
}

__global__ void k_vh_msg_dec(const int* __restrict__ EI, const float* __restrict__ NVb,
                             const float* __restrict__ AEV, const float* __restrict__ EVD,
                             const float* __restrict__ wh, float* __restrict__ vh,
                             unsigned short* __restrict__ A) {
  long idx = (long)blockIdx.x * blockDim.x + threadIdx.x;
  int h = (int)(idx % 33); long e = idx / 33;
  if (e >= NE) return;
  int src = EI[e], dst = EI[NE + e];
  const float* vb = (src < dst) ? NVb : AEV;
  float x = 0.f, y = 0.f, z = 0.f;
  for (int k = 0; k < 33; ++k) {
    const float* p;
    if (k < 16)       p = vb + (size_t)src * 48 + k * 3;
    else if (k == 16) p = EVD + (size_t)e * 3;
    else              p = vb + (size_t)dst * 48 + (k - 17) * 3;
    float w = wh[k * 33 + h];
    x = fmaf(p[0], w, x); y = fmaf(p[1], w, y); z = fmaf(p[2], w, z);
  }
  float* o = vh + ((size_t)e * 33 + h) * 3;
  o[0] = x; o[1] = y; o[2] = z;
  A[(size_t)e * 352 + 292 + h] = f2bf(sqrtf(fmaxf(x * x + y * y + z * z, 1e-8f)));
}

// ---------------- attention ----------------
__global__ void k_logits(const float* __restrict__ ms, const float* __restrict__ watt,
                         float* __restrict__ lg) {
  long idx = (long)blockIdx.x * blockDim.x + threadIdx.x;
  int h = (int)(idx & 3); long ec = idx >> 2;
  if (ec >= EC) return;
  const float* sr = ms + (size_t)ec * 128;
  float d = 0.f;
  for (int i = 0; i < 128; ++i) d = fmaf(sr[i], watt[i * 4 + h], d);
  lg[idx] = d;
}
__global__ void k_segmax(const int* __restrict__ EI, const float* __restrict__ lg,
                         float* __restrict__ mx) {
  long idx = (long)blockIdx.x * blockDim.x + threadIdx.x;
  int h = (int)(idx & 3); long ec = idx >> 2;
  if (ec >= EC) return;
  long e = ec >> 1; int cf = (int)(ec & 1);
  int dst = EI[NE + e];
  atomicMaxF(&mx[((size_t)dst * 2 + cf) * 4 + h], lg[idx]);
}
__global__ void k_alpha(const int* __restrict__ EI, float* __restrict__ lg,
                        const float* __restrict__ mx, float* __restrict__ den) {
  long idx = (long)blockIdx.x * blockDim.x + threadIdx.x;
  int h = (int)(idx & 3); long ec = idx >> 2;
  if (ec >= EC) return;
  long e = ec >> 1; int cf = (int)(ec & 1);
  int dst = EI[NE + e];
  float a = __expf(lg[idx] - mx[((size_t)dst * 2 + cf) * 4 + h]);
  lg[idx] = a;
  atomicAdd(&den[((size_t)dst * 2 + cf) * 4 + h], a);
}
__global__ void k_alpha_div(const int* __restrict__ EI, float* __restrict__ lg,
                            const float* __restrict__ den) {
  long idx = (long)blockIdx.x * blockDim.x + threadIdx.x;
  int h = (int)(idx & 3); long ec = idx >> 2;
  if (ec >= EC) return;
  long e = ec >> 1; int cf = (int)(ec & 1);
  int dst = EI[NE + e];
  lg[idx] /= (den[((size_t)dst * 2 + cf) * 4 + h] + 1e-8f);
}
// alpha-weighted scatter-add of (m_s,m_v) straight into the residual stream
__global__ void k_att_agg(const int* __restrict__ EI, const float* __restrict__ ms,
                          const float* __restrict__ mv, const float* __restrict__ lg,
                          float* __restrict__ sc, float* __restrict__ vc) {
  long idx = (long)blockIdx.x * blockDim.x + threadIdx.x;
  int f = (int)(idx % 176); long ec = idx / 176;
  if (ec >= EC) return;
  long e = ec >> 1; int cf = (int)(ec & 1);
  size_t dstc = (size_t)EI[NE + e] * 2 + cf;
  if (f < 128) {
    atomicAdd(&sc[dstc * 128 + f], ms[(size_t)ec * 128 + f] * lg[ec * 4 + (f >> 5)]);
  } else {
    int j = f - 128;
    atomicAdd(&vc[dstc * 48 + j], mv[(size_t)ec * 48 + j] * lg[ec * 4 + j / 12]);
  }
}

// ---------------- decoder mean aggregation ----------------
__global__ void k_cnt(const int* __restrict__ EI, float* __restrict__ cnt) {
  long e = (long)blockIdx.x * blockDim.x + threadIdx.x;
  if (e < NE) atomicAdd(&cnt[EI[NE + e]], 1.0f);
}
__global__ void k_clamp1(float* __restrict__ p, long n) {
  long i = (long)blockIdx.x * blockDim.x + threadIdx.x;
  if (i < n) p[i] = fmaxf(p[i], 1.f);
}
__global__ void k_dec_agg(const int* __restrict__ EI, const float* __restrict__ ms,
                          const float* __restrict__ mv, float* __restrict__ aggs,
                          float* __restrict__ aggv) {
  long idx = (long)blockIdx.x * blockDim.x + threadIdx.x;
  int f = (int)(idx % 176); long e = idx / 176;
  if (e >= NE) return;
  size_t dst = (size_t)EI[NE + e];
  if (f < 128) atomicAdd(&aggs[dst * 128 + f], ms[(size_t)e * 128 + f]);
  else { int j = f - 128; atomicAdd(&aggv[dst * 48 + j], mv[(size_t)e * 48 + j]); }
}
__global__ void k_mean_add(float* __restrict__ sD, float* __restrict__ vD,
                           const float* __restrict__ aggs, const float* __restrict__ aggv,
                           const float* __restrict__ cnt) {
  long idx = (long)blockIdx.x * blockDim.x + threadIdx.x;
  int f = (int)(idx % 176); long n = idx / 176;
  if (n >= NN) return;
  float ic = 1.f / cnt[n];
  if (f < 128) sD[(size_t)n * 128 + f] += aggs[(size_t)n * 128 + f] * ic;
  else { int j = f - 128; vD[(size_t)n * 48 + j] += aggv[(size_t)n * 48 + j] * ic; }
}

// ---------------- conformer pooling ----------------
__global__ void k_pool_nodes(const float* __restrict__ sc, const float* __restrict__ vc,
                             const float* __restrict__ mask, float* __restrict__ enc_s,
                             float* __restrict__ enc_v, float* __restrict__ sD,
                             float* __restrict__ vD) {
  long idx = (long)blockIdx.x * blockDim.x + threadIdx.x;
  int f = (int)(idx % 176); long n = idx / 176;
  if (n >= NN) return;
  float m0 = mask[n * 2], m1 = mask[n * 2 + 1];
  float inc = 1.f / (m0 + m1);
  if (f < 128) {
    float v = (sc[((size_t)n * 2) * 128 + f] * m0 + sc[((size_t)n * 2 + 1) * 128 + f] * m1) * inc;
    enc_s[(size_t)n * 128 + f] = v; sD[(size_t)n * 128 + f] = v;
  } else {
    int j = f - 128;
    float v = (vc[((size_t)n * 2) * 48 + j] * m0 + vc[((size_t)n * 2 + 1) * 48 + j] * m1) * inc;
    enc_v[(size_t)n * 48 + j] = v; vD[(size_t)n * 48 + j] = v;
  }
}
__global__ void k_pool_edges(const int* __restrict__ EI, const float* __restrict__ ES_,
                             const float* __restrict__ EV_, const float* __restrict__ mask,
                             const float* __restrict__ WsTab, const int* __restrict__ seq,
                             float* __restrict__ esd, float* __restrict__ evd) {
  long idx = (long)blockIdx.x * blockDim.x + threadIdx.x;
  int f = (int)(idx % 39); long e = idx / 39;
  if (e >= NE) return;
  int src = EI[e], dst = EI[NE + e];
  float m0 = mask[src * 2], m1 = mask[src * 2 + 1];
  float inc = 1.f / (m0 + m1);
  if (f < 32)
    esd[(size_t)e * 36 + f] = (ES_[((size_t)e * 2) * 32 + f] * m0 + ES_[((size_t)e * 2 + 1) * 32 + f] * m1) * inc;
  else if (f < 36) {
    int o = f - 32;
    esd[(size_t)e * 36 + f] = (src < dst) ? WsTab[seq[src] * 4 + o] : 0.f;
  } else {
    int c = f - 36;
    evd[(size_t)e * 3 + c] = (EV_[((size_t)e * 2) * 3 + c] * m0 + EV_[((size_t)e * 2 + 1) * 3 + c] * m1) * inc;
  }
}

// ---------------- output head ----------------
__global__ void k_out_head(const float* __restrict__ sD, const float* __restrict__ vD,
                           const float* __restrict__ wh, const float* __restrict__ ws,
                           const float* __restrict__ bs, float* __restrict__ out) {
  long n = (long)blockIdx.x * blockDim.x + threadIdx.x;
  if (n >= NN) return;
  float a0 = bs[0], a1 = bs[1], a2 = bs[2], a3 = bs[3];
  const float* sr = sD + (size_t)n * 128;
  for (int i = 0; i < 128; ++i) {
    float si = sr[i]; const float* w = ws + i * 4;
    a0 = fmaf(si, w[0], a0); a1 = fmaf(si, w[1], a1);
    a2 = fmaf(si, w[2], a2); a3 = fmaf(si, w[3], a3);
  }
  const float* vr = vD + (size_t)n * 48;
  for (int h = 0; h < 16; ++h) {
    float x = 0.f, y = 0.f, z = 0.f;
    for (int k = 0; k < 16; ++k) {
      float w = wh[k * 16 + h];
      x = fmaf(vr[k * 3], w, x); y = fmaf(vr[k * 3 + 1], w, y); z = fmaf(vr[k * 3 + 2], w, z);
    }
    float nm = sqrtf(fmaxf(x * x + y * y + z * z, 1e-8f));
    const float* w = ws + (128 + h) * 4;
    a0 = fmaf(nm, w[0], a0); a1 = fmaf(nm, w[1], a1);
    a2 = fmaf(nm, w[2], a2); a3 = fmaf(nm, w[3], a3);
  }
  float* o = out + n * 4;
  o[0] = a0; o[1] = a1; o[2] = a2; o[3] = a3;
}

// =======================================================================================
#define L1(kern, total, ...)                                                             \
  kern<<<dim3((unsigned)(((long long)(total) + 255) / 256)), dim3(256), 0, stream>>>(__VA_ARGS__)

struct GvpW { const float *bs, *bsv, *wh, *ws, *wsv, *wv; };
struct LnW  { const float *b, *g; };
struct LayerW { GvpW ff0, ff1, m0, m1, m2; LnW n0, n1; const float* watt; };

extern "C" void kernel_launch(void* const* d_in, const int* in_sizes, int n_in,
                              void* d_out, int out_size, void* d_ws, size_t ws_size,
                              hipStream_t stream) {
  (void)in_sizes; (void)n_in; (void)out_size; (void)ws_size;
  // ---- inputs (jax pytree flatten order: sorted dict keys) ----
  const int*   EI     = (const int*)d_in[0];    // edge_index (2, NE)
  const float* edge_s = (const float*)d_in[1];
  const float* edge_v = (const float*)d_in[2];
  const float* mask   = (const float*)d_in[3];
  const float* node_s = (const float*)d_in[4];
  const float* node_v = (const float*)d_in[5];
  int pi = 6;
  auto F = [&]() { return (const float*)d_in[pi++]; };
  auto getGvp = [&](bool gate) {
    GvpW w{}; w.bs = F(); if (gate) w.bsv = F();
    w.wh = F(); w.ws = F(); if (gate) { w.wsv = F(); w.wv = F(); }
    return w;
  };
  auto getLn = [&]() { LnW l; l.b = F(); l.g = F(); return l; };
  GvpW We = getGvp(true);
  LnW  We_ln = getLn();
  const float* Wout_bs = F(); const float* Wout_wh = F(); const float* Wout_ws = F();
  const float* WsTab = F();
  GvpW Wv = getGvp(true);
  LnW  Wv_ln = getLn();
  LayerW decL[3], encL[3];
  for (int l = 0; l < 3; ++l) {
    LayerW L{};
    L.ff0 = getGvp(true); L.ff1 = getGvp(true);
    L.m0 = getGvp(true); L.m1 = getGvp(true); L.m2 = getGvp(true);
    L.n0 = getLn(); L.n1 = getLn(); L.watt = nullptr;
    decL[l] = L;
  }
  for (int l = 0; l < 3; ++l) {
    LayerW L{};
    L.ff0 = getGvp(true); L.ff1 = getGvp(true);
    L.m0 = getGvp(true); L.m1 = getGvp(true); L.m2 = getGvp(true);
    L.n0 = getLn(); L.n1 = getLn(); L.watt = F();
    encL[l] = L;
  }
  const int* seq = (const int*)d_in[pi++];
  float* out = (float*)d_out;

  // ---- workspace arena ----
  char* base = (char*)d_ws;
  size_t cur = 0;
  auto aF = [&](size_t n) { float* p = (float*)(base + cur); cur = (cur + n * 4 + 255) & ~(size_t)255; return p; };
  auto aH = [&](size_t n) { unsigned short* p = (unsigned short*)(base + cur); cur = (cur + n * 2 + 255) & ~(size_t)255; return p; };
  float* S_CUR = aF((size_t)NC * 128);
  float* V_CUR = aF((size_t)NC * 48);
  float* NS    = aF((size_t)NC * 128);
  float* NV    = aF((size_t)NC * 48);
  float* ENC_S = aF((size_t)NN * 128);
  float* ENC_V = aF((size_t)NN * 48);
  float* AE_S  = aF((size_t)NN * 128);
  float* AE_V  = aF((size_t)NN * 48);
  float* SD    = aF((size_t)NN * 128);
  float* VD    = aF((size_t)NN * 48);
  float* ES    = aF((size_t)EC * 32);
  float* EV    = aF((size_t)EC * 3);
  float* ESD   = aF((size_t)NE * 36);
  float* EVD   = aF((size_t)NE * 3);
  float* CNT   = aF((size_t)NN);
  float* LOG   = aF((size_t)EC * 4);
  float* SEGMX = aF((size_t)NC * 4);
  float* DEN   = aF((size_t)NC * 4);
  float* AGGS  = aF((size_t)NN * 128);
  float* AGGV  = aF((size_t)NN * 48);
  unsigned short* A1 = aH((size_t)EC * 352);
  unsigned short* A2 = aH((size_t)EC * 160);
  unsigned short* WT = aH((size_t)512 * 544);
  float* VHB = aF((size_t)EC * 99);   // vh scratch (max H=33)
  float* MS  = aF((size_t)EC * 128);  // scalar stream scratch (also holds NC*512)
  float* MV  = aF((size_t)EC * 48);   // vector stream scratch

  auto gemm = [&](const unsigned short* A, int Kp, const float* ws, const float* bs,
                  int K, int N, int M, float* C) {
    long t = (long)N * Kp;
    k_w2bf<<<dim3((unsigned)((t + 255) / 256)), 256, 0, stream>>>(ws, WT, K, N, Kp);
    if ((N & 63) == 0) {
      dim3 g((unsigned)((M + 15) / 16), (unsigned)(N / 64));
      k_gemm_bf16<4><<<g, 32, 0, stream>>>(A, WT, bs, C, M, N, Kp);
    } else {  // N == 32
      dim3 g((unsigned)((M + 15) / 16), (unsigned)(N / 32));
      k_gemm_bf16<2><<<g, 32, 0, stream>>>(A, WT, bs, C, M, N, Kp);
    }
  };

  // ============ input embeddings ============
  // node: ln + Wv
  L1(k_ln, (long)NC * 32, node_s, node_v, Wv_ln.g, Wv_ln.b, NS, NV, (long)NC, 64, 4);
  L1(k_build_A, (long)NC * 96, NS, A2, 64, 96, (long)NC);
  L1(k_vh_norms, (long)NC * 16, NV, Wv.wh, VHB, A2, 64, 96, 4, 16, (long)NC);
  gemm(A2, 96, Wv.ws, Wv.bs, 80, 128, NC, S_CUR);
  L1(k_gate, (long)NC * 16, S_CUR, VHB, Wv.wv, Wv.wsv, Wv.bsv, V_CUR, 128, 16, 16, (long)NC);
  // edge: ln + We
  L1(k_ln, (long)EC * 32, edge_s, edge_v, We_ln.g, We_ln.b, MS, MV, (long)EC, 32, 1);
  L1(k_build_A, (long)EC * 64, MS, A2, 32, 64, (long)EC);
  L1(k_vh_norms, (long)EC * 1, MV, We.wh, VHB, A2, 32, 64, 1, 1, (long)EC);
  gemm(A2, 64, We.ws, We.bs, 33, 32, EC, ES);
  L1(k_gate, (long)EC * 1, ES, VHB, We.wv, We.wsv, We.bsv, EV, 32, 1, 1, (long)EC);

  // ============ encoder layers ============
  for (int l = 0; l < 3; ++l) {
    const LayerW& L = encL[l];
    L1(k_ln, (long)NC * 32, S_CUR, V_CUR, L.n0.g, L.n0.b, NS, NV, (long)NC, 128, 16);
    // message GVP0 (K=321)
    L1(k_scat_enc, (long)EC * 352, EI, NS, ES, A1);
    L1(k_vh_msg_enc, (long)EC * 33, EI, NV, EV, L.m0.wh, VHB, A1);
    gemm(A1, 352, L.m0.ws, L.m0.bs, 321, 128, EC, MS);
    L1(k_gate, (long)EC * 16, MS, VHB, L.m0.wv, L.m0.wsv, L.m0.bsv, MV, 128, 33, 16, (long)EC);
    L1(k_silu, (long)EC * 128, MS, (long)EC * 128);
    // message GVP1 (K=144)
    L1(k_build_A, (long)EC * 160, MS, A2, 128, 160, (long)EC);
    L1(k_vh_norms, (long)EC * 16, MV, L.m1.wh, VHB, A2, 128, 160, 16, 16, (long)EC);
    gemm(A2, 160, L.m1.ws, L.m1.bs, 144, 128, EC, MS);
    L1(k_gate, (long)EC * 16, MS, VHB, L.m1.wv, L.m1.wsv, L.m1.bsv, MV, 128, 16, 16, (long)EC);
    L1(k_silu, (long)EC * 128, MS, (long)EC * 128);
    // message GVP2 (K=144, no act)
    L1(k_build_A, (long)EC * 160, MS, A2, 128, 160, (long)EC);
    L1(k_vh_norms, (long)EC * 16, MV, L.m2.wh, VHB, A2, 128, 160, 16, 16, (long)EC);
    gemm(A2, 160, L.m2.ws, L.m2.bs, 144, 128, EC, MS);
    L1(k_gate, (long)EC * 16, MS, VHB, L.m2.wv, L.m2.wsv, L.m2.bsv, MV, 128, 16, 16, (long)EC);
    // attention + residual scatter
    L1(k_logits, (long)EC * 4, MS, L.watt, LOG);
    L1(k_fill, (long)NC * 4, SEGMX, -1e30f, (long)NC * 4);
    L1(k_segmax, (long)EC * 4, EI, LOG, SEGMX);
    L1(k_fill, (long)NC * 4, DEN, 0.f, (long)NC * 4);
    L1(k_alpha, (long)EC * 4, EI, LOG, SEGMX, DEN);
    L1(k_alpha_div, (long)EC * 4, EI, LOG, DEN);
    L1(k_att_agg, (long)EC * 176, EI, MS, MV, LOG, S_CUR, V_CUR);
    // feed-forward
    L1(k_ln, (long)NC * 32, S_CUR, V_CUR, L.n1.g, L.n1.b, NS, NV, (long)NC, 128, 16);
    L1(k_build_A, (long)NC * 160, NS, A2, 128, 160, (long)NC);
    L1(k_vh_norms, (long)NC * 32, NV, L.ff0.wh, VHB, A2, 128, 160, 16, 32, (long)NC);
    gemm(A2, 160, L.ff0.ws, L.ff0.bs, 160, 512, NC, MS);
    L1(k_gate, (long)NC * 32, MS, VHB, L.ff0.wv, L.ff0.wsv, L.ff0.bsv, MV, 512, 32, 32, (long)NC);
    L1(k_silu, (long)NC * 512, MS, (long)NC * 512);
    L1(k_build_A, (long)NC * 544, MS, A1, 512, 544, (long)NC);
    L1(k_vh_norms, (long)NC * 32, MV, L.ff1.wh, VHB, A1, 512, 544, 32, 32, (long)NC);
    gemm(A1, 544, L.ff1.ws, L.ff1.bs, 544, 128, NC, MS);
    L1(k_gate, (long)NC * 16, MS, VHB, L.ff1.wv, L.ff1.wsv, L.ff1.bsv, MV, 128, 32, 16, (long)NC);
    L1(k_axpy, (long)NC * 128, S_CUR, MS, (long)NC * 128);
    L1(k_axpy, (long)NC * 48, V_CUR, MV, (long)NC * 48);
  }

  // ============ conformer pooling + decoder edge features ============
  L1(k_pool_nodes, (long)NN * 176, S_CUR, V_CUR, mask, ENC_S, ENC_V, SD, VD);
  L1(k_pool_edges, (long)NE * 39, EI, ES, EV, mask, WsTab, seq, ESD, EVD);
  L1(k_fill, (long)NN, CNT, 0.f, (long)NN);
  L1(k_cnt, (long)NE, EI, CNT);
  L1(k_clamp1, (long)NN, CNT, (long)NN);

  // ============ decoder layers ============
  for (int l = 0; l < 3; ++l) {
    const LayerW& L = decL[l];
    L1(k_ln, (long)NN * 32, SD, VD, L.n0.g, L.n0.b, NS, NV, (long)NN, 128, 16);
    L1(k_ln, (long)NN * 32, ENC_S, ENC_V, L.n0.g, L.n0.b, AE_S, AE_V, (long)NN, 128, 16);
    // message GVP0 (K=325)
    L1(k_scat_dec, (long)NE * 352, EI, NS, AE_S, ESD, A1);
    L1(k_vh_msg_dec, (long)NE * 33, EI, NV, AE_V, EVD, L.m0.wh, VHB, A1);
    gemm(A1, 352, L.m0.ws, L.m0.bs, 325, 128, NE, MS);
    L1(k_gate, (long)NE * 16, MS, VHB, L.m0.wv, L.m0.wsv, L.m0.bsv, MV, 128, 33, 16, (long)NE);
    L1(k_silu, (long)NE * 128, MS, (long)NE * 128);
    // message GVP1
    L1(k_build_A, (long)NE * 160, MS, A2, 128, 160, (long)NE);
    L1(k_vh_norms, (long)NE * 16, MV, L.m1.wh, VHB, A2, 128, 160, 16, 16, (long)NE);
    gemm(A2, 160, L.m1.ws, L.m1.bs, 144, 128, NE, MS);
    L1(k_gate, (long)NE * 16, MS, VHB, L.m1.wv, L.m1.wsv, L.m1.bsv, MV, 128, 16, 16, (long)NE);
    L1(k_silu, (long)NE * 128, MS, (long)NE * 128);
    // message GVP2
    L1(k_build_A, (long)NE * 160, MS, A2, 128, 160, (long)NE);
    L1(k_vh_norms, (long)NE * 16, MV, L.m2.wh, VHB, A2, 128, 160, 16, 16, (long)NE);
    gemm(A2, 160, L.m2.ws, L.m2.bs, 144, 128, NE, MS);
    L1(k_gate, (long)NE * 16, MS, VHB, L.m2.wv, L.m2.wsv, L.m2.bsv, MV, 128, 16, 16, (long)NE);
    // mean aggregation + residual
    L1(k_fill, (long)NN * 128, AGGS, 0.f, (long)NN * 128);
    L1(k_fill, (long)NN * 48, AGGV, 0.f, (long)NN * 48);
    L1(k_dec_agg, (long)NE * 176, EI, MS, MV, AGGS, AGGV);
    L1(k_mean_add, (long)NN * 176, SD, VD, AGGS, AGGV, CNT);
    // feed-forward
    L1(k_ln, (long)NN * 32, SD, VD, L.n1.g, L.n1.b, NS, NV, (long)NN, 128, 16);
    L1(k_build_A, (long)NN * 160, NS, A2, 128, 160, (long)NN);
    L1(k_vh_norms, (long)NN * 32, NV, L.ff0.wh, VHB, A2, 128, 160, 16, 32, (long)NN);
    gemm(A2, 160, L.ff0.ws, L.ff0.bs, 160, 512, NN, MS);
    L1(k_gate, (long)NN * 32, MS, VHB, L.ff0.wv, L.ff0.wsv, L.ff0.bsv, MV, 512, 32, 32, (long)NN);
    L1(k_silu, (long)NN * 512, MS, (long)NN * 512);
    L1(k_build_A, (long)NN * 544, MS, A1, 512, 544, (long)NN);
    L1(k_vh_norms, (long)NN * 32, MV, L.ff1.wh, VHB, A1, 512, 544, 32, 32, (long)NN);
    gemm(A1, 544, L.ff1.ws, L.ff1.bs, 544, 128, NN, MS);
    L1(k_gate, (long)NN * 16, MS, VHB, L.ff1.wv, L.ff1.wsv, L.ff1.bsv, MV, 128, 32, 16, (long)NN);
    L1(k_axpy, (long)NN * 128, SD, MS, (long)NN * 128);
    L1(k_axpy, (long)NN * 48, VD, MV, (long)NN * 48);
  }

  // ============ output head ============
  L1(k_out_head, (long)NN, SD, VD, Wout_wh, Wout_ws, Wout_bs, out);
}
